// ImageFusion_78597901517117
// MI455X (gfx1250) — matrix operand order repaired and checked by reading, not compiled, not run on hardware
//
#include <hip/hip_runtime.h>

// ---------------------------------------------------------------------------
// Types for CDNA5 WMMA (wave32): v_wmma_f32_16x16x32_bf16
// ---------------------------------------------------------------------------
typedef __bf16 bf16_t;
typedef bf16_t v16bf __attribute__((ext_vector_type(16)));
typedef bf16_t v8bf  __attribute__((ext_vector_type(8)));
typedef float  v8f   __attribute__((ext_vector_type(8)));

#define LDS_STRIDE 40   // ushorts per LDS row/col (80B) -> spreads LDS banks

__device__ __forceinline__ unsigned short f2bf(float f) {
  unsigned u = __float_as_uint(f);
  unsigned r = u + 0x7fffu + ((u >> 16) & 1u);   // round-to-nearest-even
  return (unsigned short)(r >> 16);
}
__device__ __forceinline__ float bf2f(unsigned short h) {
  return __uint_as_float(((unsigned)h) << 16);
}
// LDS byte offset of a __shared__ pointer (flat LDS addr truncates to [31:0])
__device__ __forceinline__ unsigned lds_off32(const void* p) {
  return (unsigned)(size_t)p;
}

// ---------------------------------------------------------------------------
// Stage 1: LayerNorm over C=32 + im2col into bf16 matrix Xc[2048 x 2048]
//   row r = b*1024 + gh*32 + gw ; col = (p0*8+p1)*32 + c
// ---------------------------------------------------------------------------
__global__ __launch_bounds__(256) void k_ln_im2col(
    const float* __restrict__ x, const float* __restrict__ sc,
    const float* __restrict__ bi, unsigned short* __restrict__ Xc)
{
  int pix = blockIdx.x * 256 + threadIdx.x;                 // 0..131071
  const float4* px = reinterpret_cast<const float4*>(x + (size_t)pix * 32);
  float v[32];
#pragma unroll
  for (int i = 0; i < 8; ++i) {
    float4 f = px[i];
    v[4*i+0]=f.x; v[4*i+1]=f.y; v[4*i+2]=f.z; v[4*i+3]=f.w;
  }
  float mu = 0.f;
#pragma unroll
  for (int c = 0; c < 32; ++c) mu += v[c];
  mu *= (1.f/32.f);
  float var = 0.f;
#pragma unroll
  for (int c = 0; c < 32; ++c) { float d = v[c]-mu; var += d*d; }
  var *= (1.f/32.f);
  float rs = rsqrtf(var + 1e-6f);
  alignas(16) unsigned short o[32];
#pragma unroll
  for (int c = 0; c < 32; ++c)
    o[c] = f2bf((v[c]-mu)*rs*sc[c] + bi[c]);
  int b = pix >> 16, h = (pix >> 8) & 255, w = pix & 255;
  int r = b*1024 + (h>>3)*32 + (w>>3);
  int colb = ((h&7)*8 + (w&7)) * 32;
  uint4* dst = reinterpret_cast<uint4*>(Xc + (size_t)r*2048 + colb);
  const uint4* so = reinterpret_cast<const uint4*>(o);
  dst[0]=so[0]; dst[1]=so[1]; dst[2]=so[2]; dst[3]=so[3];
}

// ---------------------------------------------------------------------------
// GEMM: C[M x N] = A(bf16)[M x K] * W(f32->bf16)[K x N] + bias, WMMA bf16.
// WG = 256 threads = 8 waves (2x4), tile 128x128, K-chunk 32.
// A tile staged with CDNA5 async global->LDS DMA (no VGPR bounce).
// mode 0: scatter into [patch*8+head][c][s] bf16 layout (QKV projections)
// mode 1: bias + residual + NHWC scatter to fp32 output (final projection)
// ---------------------------------------------------------------------------
__global__ __launch_bounds__(256) void k_gemm_bf16(
    const unsigned short* __restrict__ A, const float* __restrict__ W,
    const float* __restrict__ bias, int M, int N, int K, int mode,
    unsigned short* __restrict__ dstG, float* __restrict__ outF,
    const float* __restrict__ xres)
{
  (void)M;
  __shared__ __align__(64) unsigned short Asm[128 * LDS_STRIDE];
  __shared__ __align__(64) unsigned short Bsm[128 * LDS_STRIDE];
  const int t = threadIdx.x;
  const int lane = t & 31;
  const int wid  = t >> 5;
  const int wm = wid >> 2;           // 0..1 -> M offset wm*64
  const int wn = wid & 3;            // 0..3 -> N offset wn*32
  const int nBase   = blockIdx.x * 128;
  const int rowBase = blockIdx.y * 128;
  const int h = lane >> 4, l15 = lane & 15;

  // Per-thread A-copy geometry (2 chunks of 16B; 512 chunks of 16B total)
  const int rowA0 = t >> 2,           qA0 = t & 3;
  const int rowA1 = (t + 256) >> 2,   qA1 = t & 3;
  const unsigned ldsA0 = lds_off32(&Asm[rowA0 * LDS_STRIDE + qA0 * 8]);
  const unsigned ldsA1 = lds_off32(&Asm[rowA1 * LDS_STRIDE + qA1 * 8]);
  const unsigned short* Arow0 = A + (size_t)(rowBase + rowA0) * K + qA0 * 8;
  const unsigned short* Arow1 = A + (size_t)(rowBase + rowA1) * K + qA1 * 8;

  v8f acc[4][2];
#pragma unroll
  for (int i=0;i<4;++i)
#pragma unroll
    for (int j=0;j<2;++j)
#pragma unroll
      for (int e=0;e<8;++e) acc[i][j][e] = 0.f;

  for (int k0 = 0; k0 < K; k0 += 32) {
    // ---- stage A tile via async global->LDS DMA (bf16, no conversion) ----
    asm volatile("global_load_async_to_lds_b128 %0, %1, off"
                 :: "v"(ldsA0), "v"(Arow0 + k0) : "memory");
    asm volatile("global_load_async_to_lds_b128 %0, %1, off"
                 :: "v"(ldsA1), "v"(Arow1 + k0) : "memory");

    // ---- stage B tile (32 K x 128 N) f32 -> bf16, transposed (col-major) ----
    // batch all global loads first, then convert + store (no per-load stalls)
    const float* Wp = W + (size_t)k0 * N + nBase;
    float2 wlo[4], whi[4];
#pragma unroll
    for (int i = 0; i < 4; ++i) {
      int idx = i*256 + t;                 // 0..1023
      int c2  = (idx & 63) * 2;            // column pair 0,2,..,126
      int kp  = idx >> 6;                  // K pair 0..15
      wlo[i] = *reinterpret_cast<const float2*>(Wp + (unsigned)(kp*2    )*N + c2);
      whi[i] = *reinterpret_cast<const float2*>(Wp + (unsigned)(kp*2 + 1)*N + c2);
    }
    if (k0 + 32 < K)
      __builtin_prefetch(Wp + (unsigned)32*N + (t & 127), 0, 0);
#pragma unroll
    for (int i = 0; i < 4; ++i) {
      int idx = i*256 + t;
      int c2  = (idx & 63) * 2;
      int kp  = idx >> 6;
      unsigned p0 = (unsigned)f2bf(wlo[i].x) | ((unsigned)f2bf(whi[i].x) << 16);
      unsigned p1 = (unsigned)f2bf(wlo[i].y) | ((unsigned)f2bf(whi[i].y) << 16);
      reinterpret_cast<unsigned*>(&Bsm[c2 * LDS_STRIDE])[kp] = p0;
      reinterpret_cast<unsigned*>(&Bsm[(c2+1) * LDS_STRIDE])[kp] = p1;
    }

    // async A must be resident in LDS before the cross-wave barrier
    asm volatile("s_wait_asynccnt 0x0" ::: "memory");
    __syncthreads();

    // ---- WMMA fragments from LDS per ISA layouts ----
    v16bf af[4], bfr[2];
#pragma unroll
    for (int i=0;i<4;++i){
      int row = wm*64 + i*16 + l15;
      // A lane holds K = {8h..8h+7} then {16+8h..16+8h+7}
      v8bf lo = *reinterpret_cast<const v8bf*>(&Asm[row*LDS_STRIDE + 8*h]);
      v8bf hi = *reinterpret_cast<const v8bf*>(&Asm[row*LDS_STRIDE + 16 + 8*h]);
#pragma unroll
      for (int e=0;e<8;++e){ af[i][e]=lo[e]; af[i][8+e]=hi[e]; }
    }
#pragma unroll
    for (int j=0;j<2;++j){
      // B lane holds column l15, K = {8h..8h+7} then {16+8h..16+8h+7}
      int col = wn*32 + j*16 + l15;
      v8bf blo = *reinterpret_cast<const v8bf*>(&Bsm[col*LDS_STRIDE + 8*h]);
      v8bf bhi = *reinterpret_cast<const v8bf*>(&Bsm[col*LDS_STRIDE + 16 + 8*h]);
#pragma unroll
      for (int e=0;e<8;++e){ bfr[j][e]=blo[e]; bfr[j][8+e]=bhi[e]; }
    }
#pragma unroll
    for (int i=0;i<4;++i)
#pragma unroll
      for (int j=0;j<2;++j)
        acc[i][j] = __builtin_amdgcn_wmma_f32_16x16x32_bf16(
            false, af[i], false, bfr[j], (short)0, acc[i][j], false, false);
    __syncthreads();
  }

  // ---- epilogue: C/D layout m = e + 8h, n = lane&15 ----
#pragma unroll
  for (int i=0;i<4;++i) {
#pragma unroll
    for (int j=0;j<2;++j) {
      int n = nBase + wn*32 + j*16 + l15;
      float bv = bias[n];
#pragma unroll
      for (int e=0;e<8;++e) {
        int r = rowBase + wm*64 + i*16 + 8*h + e;
        float val = acc[i][j][e] + bv;
        if (mode == 0) {
          // f = c*512 + s*8 + head  ->  [r*8+head][c*64+s]
          int head = n & 7, s = (n >> 3) & 63, c = n >> 9;
          dstG[((size_t)r*8 + head)*2048 + (size_t)c*64 + s] = f2bf(val);
        } else {
          // g = c*64 + p0*8 + p1 ; r = b*1024 + gh*32 + gw
          int c = n >> 6, pp = n & 63;
          int b = r >> 10, gh = (r >> 5) & 31, gw = r & 31;
          int hh = gh*8 + (pp >> 3), ww = gw*8 + (pp & 7);
          size_t o = (((size_t)b*256 + hh)*256 + ww)*32 + c;
          outF[o] = val + xres[o];
        }
      }
    }
  }
}

// ---------------------------------------------------------------------------
// Stage 3: per (patch, head): 8x8 circular conv (rfft2*rfft2 -> irfft2),
// LN2 over c, gate by V, write bf16 im2col matrix Og[2048 x 16384]
// (col = s*256 + head*32 + c) for the output GEMM.
// One 64-thread block per group; thread = spatial s holds all 32 channels.
// ---------------------------------------------------------------------------
__global__ __launch_bounds__(64) void k_fftgate(
    const unsigned short* __restrict__ Qg, const unsigned short* __restrict__ Kg,
    const unsigned short* __restrict__ Vg, const float* __restrict__ sc,
    const float* __restrict__ bi, unsigned short* __restrict__ Og)
{
  __shared__ float qs[2048];
  __shared__ float ks[2048];
  const int t = threadIdx.x;                       // spatial s = p0*8+p1
  const size_t base = (size_t)blockIdx.x * 2048;   // group = r*8 + head
  for (int i = t; i < 2048; i += 64) {
    qs[i] = bf2f(Qg[base + i]);
    ks[i] = bf2f(Kg[base + i]);
  }
  __syncthreads();
  const int s0 = t >> 3, s1 = t & 7;
  float acc[32];
#pragma unroll
  for (int c=0;c<32;++c) acc[c]=0.f;
  for (int a0=0;a0<8;++a0){
    for (int a1=0;a1<8;++a1){
      int a = a0*8 + a1;
      int idx = (((s0 - a0) & 7) << 3) | ((s1 - a1) & 7);
#pragma unroll
      for (int c=0;c<32;++c)
        acc[c] = fmaf(qs[c*64 + a], ks[c*64 + idx], acc[c]);
    }
  }
  float mu=0.f;
#pragma unroll
  for (int c=0;c<32;++c) mu += acc[c];
  mu *= (1.f/32.f);
  float var=0.f;
#pragma unroll
  for (int c=0;c<32;++c){ float d=acc[c]-mu; var += d*d; }
  var *= (1.f/32.f);
  float rs = rsqrtf(var + 1e-6f);
  int head = blockIdx.x & 7;
  size_t rowOff = (size_t)(blockIdx.x >> 3) * 16384 + (size_t)t*256 + head*32;
#pragma unroll
  for (int c=0;c<32;++c){
    float g = (acc[c]-mu)*rs*sc[c] + bi[c];
    float v = bf2f(Vg[base + (size_t)c*64 + t]);
    Og[rowOff + c] = f2bf(v * g);
  }
}

// ---------------------------------------------------------------------------
extern "C" void kernel_launch(void* const* d_in, const int* in_sizes, int n_in,
                              void* d_out, int out_size, void* d_ws, size_t ws_size,
                              hipStream_t stream)
{
  (void)in_sizes; (void)n_in; (void)out_size; (void)ws_size;
  const float* x   = (const float*)d_in[0];
  const float* l1s = (const float*)d_in[1];
  const float* l1b = (const float*)d_in[2];
  const float* Wq  = (const float*)d_in[3];
  const float* bq  = (const float*)d_in[4];
  const float* Wk  = (const float*)d_in[5];
  const float* bk  = (const float*)d_in[6];
  const float* Wv  = (const float*)d_in[7];
  const float* bv  = (const float*)d_in[8];
  const float* l2s = (const float*)d_in[9];
  const float* l2b = (const float*)d_in[10];
  const float* Wo  = (const float*)d_in[11];
  const float* bo  = (const float*)d_in[12];
  float* out = (float*)d_out;

  char* w = (char*)d_ws;
  unsigned short* Xc = (unsigned short*)(w);                       // 8 MiB
  unsigned short* Qg = (unsigned short*)(w + (8ull<<20));          // 64 MiB
  unsigned short* Kg = (unsigned short*)(w + (8ull<<20) + (64ull<<20));
  unsigned short* Vg = (unsigned short*)(w + (8ull<<20) + (128ull<<20));
  unsigned short* Og = (unsigned short*)(w + (8ull<<20) + (192ull<<20));

  k_ln_im2col<<<512, 256, 0, stream>>>(x, l1s, l1b, Xc);

  dim3 g1(128, 16);   // N tiles x M tiles
  k_gemm_bf16<<<g1, 256, 0, stream>>>(Xc, Wq, bq, 2048, 16384, 2048, 0, Qg, nullptr, nullptr);
  k_gemm_bf16<<<g1, 256, 0, stream>>>(Xc, Wk, bk, 2048, 16384, 2048, 0, Kg, nullptr, nullptr);
  k_gemm_bf16<<<g1, 256, 0, stream>>>(Xc, Wv, bv, 2048, 16384, 2048, 0, Vg, nullptr, nullptr);

  k_fftgate<<<16384, 64, 0, stream>>>(Qg, Kg, Vg, l2s, l2b, Og);

  dim3 g2(16, 16);
  k_gemm_bf16<<<g2, 256, 0, stream>>>(Og, Wo, bo, 2048, 2048, 16384, 1, nullptr, out, x);
}
